// MultiHeadAttention_52647709114863
// MI455X (gfx1250) — matrix operand order repaired and checked by reading, not compiled, not run on hardware
//
#include <hip/hip_runtime.h>

// ---------------------------------------------------------------------------
// MHA forward for MI455X (gfx1250): bf16 WMMA everywhere, f32 accumulate.
// S=4096, D_MODEL=1024, H=16, D_K=64.
// GEMMs use double-buffered LDS fed by GLOBAL_LOAD_ASYNC_TO_LDS_B128.
// ---------------------------------------------------------------------------

typedef __attribute__((ext_vector_type(16))) __bf16 v16bf;
typedef __attribute__((ext_vector_type(8)))  float  v8f;

union Frag {
    v16bf v;
    uint4 u[2];
};

#define S_LEN   4096
#define DMODEL  1024
#define NHEADS  16
#define DK      64

// Async copy 16B global -> LDS (ASYNCcnt-tracked, bypasses VGPRs).
__device__ __forceinline__ void async_copy_b128(unsigned lds_off, const void* gaddr) {
    asm volatile("global_load_async_to_lds_b128 %0, %1, off"
                 :: "v"(lds_off), "v"(gaddr) : "memory");
}
__device__ __forceinline__ void wait_async0() {
    asm volatile("s_wait_asynccnt 0x0" ::: "memory");
}
// Low 32 bits of a flat shared pointer == LDS byte address (ISA aperture rule).
__device__ __forceinline__ unsigned lds_off32(const void* p) {
    return (unsigned)(unsigned long long)p;
}

// ------------------------------- prep kernels ------------------------------

__global__ void cvt_bf16_kernel(const float* __restrict__ in,
                                __bf16* __restrict__ out, int n) {
    int i = blockIdx.x * blockDim.x + threadIdx.x;
    if (i < n) out[i] = (__bf16)in[i];
}

// W: [1024 in][1024 out] fp32  ->  Wt: [out][in] bf16
__global__ void transpose_cvt_kernel(const float* __restrict__ W,
                                     __bf16* __restrict__ Wt) {
    int i = blockIdx.x * blockDim.x + threadIdx.x;   // over 1024*1024
    int o  = i >> 10;
    int in_ = i & 1023;
    Wt[i] = (__bf16)W[in_ * DMODEL + o];
}

// ------------------------------- GEMM kernel -------------------------------
// C[M=4096][N=1024] = A_bf16[M][K] @ Wt_bf16[N][K] + bias[N]
// Block tile 128x64, 8 waves, each wave 32x32 (2x2 WMMA tiles), BK = 32.
// Double-buffered LDS, async global->LDS staging.
// mode 0: Q -> bf16 [h][s][d], scaled by 1/8
// mode 1: K -> bf16 [h][s][d]
// mode 2: V -> bf16 transposed [n][s] (== [h][d][s])
// mode 3: O -> fp32 [s][n]  (final output)

#define BM 128
#define BN 64
#define BK 32

__global__ __launch_bounds__(256) void gemm_bf16_kernel(
    const __bf16* __restrict__ A, const __bf16* __restrict__ Bt,
    const float* __restrict__ bias, __bf16* __restrict__ outb,
    float* __restrict__ outf, int Kd, int mode, float scale)
{
    __shared__ __align__(16) __bf16 As[2][BM][BK];   // 2 x 8 KB
    __shared__ __align__(16) __bf16 Bs[2][BN][BK];   // 2 x 4 KB

    const int tid = threadIdx.x;
    const int w  = tid >> 5;
    const int l  = tid & 31;
    const int lm = l & 15;
    const int hi = l >> 4;
    const int wm = w & 3;         // 4 M-strips of 32
    const int wn = w >> 2;        // 2 N-strips of 32
    const int m0 = blockIdx.y * BM;
    const int n0 = blockIdx.x * BN;

    // per-thread staging coordinates
    const int ar0 = tid >> 2,         ac0 = (tid & 3) * 8;        // A chunk 0
    const int ar1 = (tid + 256) >> 2, ac1 = (tid & 3) * 8;        // A chunk 1
    const int br  = tid >> 2,         bc  = (tid & 3) * 8;        // B chunk

    const int NT = Kd / BK;

    // issue async stage of tile `kt` into buffer `buf` (3 async ops/thread)
    auto stage = [&](int buf, int k0) {
        async_copy_b128(lds_off32(&As[buf][ar0][ac0]),
                        &A[(size_t)(m0 + ar0) * Kd + k0 + ac0]);
        async_copy_b128(lds_off32(&As[buf][ar1][ac1]),
                        &A[(size_t)(m0 + ar1) * Kd + k0 + ac1]);
        async_copy_b128(lds_off32(&Bs[buf][br][bc]),
                        &Bt[(size_t)(n0 + br) * Kd + k0 + bc]);
    };

    v8f acc[2][2] = {};

    stage(0, 0);                                   // prologue
    for (int kt = 0; kt < NT; ++kt) {
        const int cur = kt & 1;
        wait_async0();                             // own tile-kt copies done
        __syncthreads();                           // everyone's copies visible;
                                                   // everyone done reading buf
        if (kt + 1 < NT) stage(1 - cur, (kt + 1) * BK);   // overlap next tile

        // A fragments: lane = row, K chunks {hi*8..hi*8+7, 16+hi*8..}
        Frag af[2], bf[2];
#pragma unroll
        for (int am = 0; am < 2; ++am) {
            const char* base = (const char*)&As[cur][wm * 32 + am * 16 + lm][0];
            af[am].u[0] = *(const uint4*)(base + hi * 16);
            af[am].u[1] = *(const uint4*)(base + 32 + hi * 16);
        }
        // B fragments: lane = col, K chunks {hi*8..hi*8+7, 16+hi*8..}
#pragma unroll
        for (int bn = 0; bn < 2; ++bn) {
            const char* base = (const char*)&Bs[cur][wn * 32 + bn * 16 + lm][0];
            bf[bn].u[0] = *(const uint4*)(base + hi * 16);
            bf[bn].u[1] = *(const uint4*)(base + 32 + hi * 16);
        }
#pragma unroll
        for (int am = 0; am < 2; ++am)
#pragma unroll
            for (int bn = 0; bn < 2; ++bn)
                acc[am][bn] = __builtin_amdgcn_wmma_f32_16x16x32_bf16(
                    false, af[am].v, false, bf[bn].v,
                    (short)0, acc[am][bn], false, false);
    }

    // epilogue: C tile element (lane,r) -> row m = r + 8*hi, col n = lane&15
#pragma unroll
    for (int am = 0; am < 2; ++am) {
#pragma unroll
        for (int bn = 0; bn < 2; ++bn) {
            int n = n0 + wn * 32 + bn * 16 + lm;
            float bv = bias[n];
#pragma unroll
            for (int r = 0; r < 8; ++r) {
                int m = m0 + wm * 32 + am * 16 + r + hi * 8;
                float val = (acc[am][bn][r] + bv) * scale;
                if (mode <= 1) {
                    // [h][s][d] with h = n>>6, d = n&63
                    outb[((size_t)(n >> 6) * S_LEN + m) * DK + (n & 63)] = (__bf16)val;
                } else if (mode == 2) {
                    // transposed: [n][s]
                    outb[(size_t)n * S_LEN + m] = (__bf16)val;
                } else {
                    outf[(size_t)m * DMODEL + n] = val;
                }
            }
        }
    }
}

// ----------------------------- attention kernel ----------------------------
// Flash attention, one head per blockIdx.y, 128 q-rows per block (16/wave).
// Q pre-scaled by 1/sqrt(dk). K [h][s][64], Vt [h][64][s] are L2-resident.

__global__ __launch_bounds__(256) void attn_kernel(
    const __bf16* __restrict__ Qh, const __bf16* __restrict__ Kh,
    const __bf16* __restrict__ Vt, __bf16* __restrict__ O)
{
    __shared__ __align__(16) __bf16 Pbuf[8][16][32];  // per-wave P bounce

    const int tid = threadIdx.x;
    const int w  = tid >> 5;
    const int l  = tid & 31;
    const int lm = l & 15;
    const int hi = l >> 4;
    const int h     = blockIdx.y;
    const int qbase = blockIdx.x * 128 + w * 16;

    // Q fragments (16 rows x 64 d = 2 A-frags), kept in registers
    Frag qf[2];
    {
        const char* qrow = (const char*)(Qh + ((size_t)h * S_LEN + qbase + lm) * DK);
#pragma unroll
        for (int ks = 0; ks < 2; ++ks) {
            qf[ks].u[0] = *(const uint4*)(qrow + ks * 64 + hi * 16);
            qf[ks].u[1] = *(const uint4*)(qrow + ks * 64 + 32 + hi * 16);
        }
    }

    float mrow[8], lrow[8];
#pragma unroll
    for (int r = 0; r < 8; ++r) { mrow[r] = -1e30f; lrow[r] = 0.0f; }
    v8f oacc[4] = {};

    const __bf16* Kbase = Kh + (size_t)h * S_LEN * DK;
    const __bf16* Vbase = Vt + (size_t)h * DK * S_LEN;

    for (int jt = 0; jt < S_LEN / 32; ++jt) {
        const int jbase = jt * 32;

        if (jt + 1 < S_LEN / 32) {  // L2 prefetch of next 32-key K tile
            __builtin_prefetch(Kbase + (size_t)(jbase + 32 + l) * DK, 0, 1);
        }

        // ---- S = Q K^T for 32 keys (two 16-key N-tiles) ----
        v8f sacc[2];
#pragma unroll
        for (int nt = 0; nt < 2; ++nt) {
            const char* krow =
                (const char*)(Kbase + (size_t)(jbase + nt * 16 + lm) * DK);
            Frag kf[2];
#pragma unroll
            for (int ks = 0; ks < 2; ++ks) {
                kf[ks].u[0] = *(const uint4*)(krow + ks * 64 + hi * 16);
                kf[ks].u[1] = *(const uint4*)(krow + ks * 64 + 32 + hi * 16);
            }
            v8f s = {};
            s = __builtin_amdgcn_wmma_f32_16x16x32_bf16(false, qf[0].v, false,
                    kf[0].v, (short)0, s, false, false);
            s = __builtin_amdgcn_wmma_f32_16x16x32_bf16(false, qf[1].v, false,
                    kf[1].v, (short)0, s, false, false);
            sacc[nt] = s;
        }

        // ---- online softmax (rows live in 16-lane halves) ----
#pragma unroll
        for (int r = 0; r < 8; ++r) {
            float s0 = sacc[0][r], s1 = sacc[1][r];
            float sm = fmaxf(s0, s1);
#pragma unroll
            for (int off = 8; off >= 1; off >>= 1)
                sm = fmaxf(sm, __shfl_xor(sm, off, 16));
            float mnew  = fmaxf(mrow[r], sm);
            float alpha = __expf(mrow[r] - mnew);
            mrow[r] = mnew;
            float p0 = __expf(s0 - mnew);
            float p1 = __expf(s1 - mnew);
            float ps = p0 + p1;
#pragma unroll
            for (int off = 8; off >= 1; off >>= 1)
                ps += __shfl_xor(ps, off, 16);
            lrow[r] = lrow[r] * alpha + ps;
#pragma unroll
            for (int t = 0; t < 4; ++t) oacc[t][r] *= alpha;
            int prow = r + hi * 8;
            Pbuf[w][prow][lm]      = (__bf16)p0;
            Pbuf[w][prow][16 + lm] = (__bf16)p1;
        }
        // wave-private LDS bounce: only this wave touches Pbuf[w]
        asm volatile("s_wait_dscnt 0x0" ::: "memory");
        __builtin_amdgcn_wave_barrier();

        // reload P in A-fragment layout
        Frag pf;
        {
            const char* pbase = (const char*)&Pbuf[w][lm][0];
            pf.u[0] = *(const uint4*)(pbase + hi * 16);
            pf.u[1] = *(const uint4*)(pbase + 32 + hi * 16);
        }
        __builtin_amdgcn_wave_barrier();  // keep loads before next iter's stores

        // ---- O += P @ V (4 d-tiles of 16) ----
#pragma unroll
        for (int t = 0; t < 4; ++t) {
            const char* vrow =
                (const char*)(Vbase + (size_t)(t * 16 + lm) * S_LEN + jbase);
            Frag vf;
            vf.u[0] = *(const uint4*)(vrow + hi * 16);
            vf.u[1] = *(const uint4*)(vrow + 32 + hi * 16);
            oacc[t] = __builtin_amdgcn_wmma_f32_16x16x32_bf16(
                false, pf.v, false, vf.v, (short)0, oacc[t], false, false);
        }
    }

    // ---- normalize and store merged-head bf16 [s][1024] ----
#pragma unroll
    for (int t = 0; t < 4; ++t) {
        int col = h * DK + t * 16 + lm;
#pragma unroll
        for (int r = 0; r < 8; ++r) {
            int row = qbase + r + hi * 8;
            O[(size_t)row * DMODEL + col] = (__bf16)(oacc[t][r] / lrow[r]);
        }
    }
}

// ------------------------------- launcher ----------------------------------

extern "C" void kernel_launch(void* const* d_in, const int* in_sizes, int n_in,
                              void* d_out, int out_size, void* d_ws, size_t ws_size,
                              hipStream_t stream) {
    const float* x  = (const float*)d_in[0];
    const float* Wq = (const float*)d_in[1];
    const float* bq = (const float*)d_in[2];
    const float* Wk = (const float*)d_in[3];
    const float* bk = (const float*)d_in[4];
    const float* Wv = (const float*)d_in[5];
    const float* bv = (const float*)d_in[6];
    const float* Wo = (const float*)d_in[7];
    const float* bo = (const float*)d_in[8];
    float* out = (float*)d_out;

    char* ws = (char*)d_ws;
    const size_t SZ_X = (size_t)S_LEN * DMODEL * sizeof(__bf16);   // 8 MB
    const size_t SZ_W = (size_t)DMODEL * DMODEL * sizeof(__bf16);  // 2 MB
    __bf16* xb  = (__bf16*)ws;             ws += SZ_X;
    __bf16* Wqt = (__bf16*)ws;             ws += SZ_W;
    __bf16* Wkt = (__bf16*)ws;             ws += SZ_W;
    __bf16* Wvt = (__bf16*)ws;             ws += SZ_W;
    __bf16* Wot = (__bf16*)ws;             ws += SZ_W;
    __bf16* Qh  = (__bf16*)ws;             ws += SZ_X;
    __bf16* Kh  = (__bf16*)ws;             ws += SZ_X;
    __bf16* Vt  = (__bf16*)ws;             ws += SZ_X;
    __bf16* Ab  = (__bf16*)ws;             ws += SZ_X;   // attention output (bf16)

    // 1) convert x and weights (weights transposed to [out][in])
    cvt_bf16_kernel<<<(S_LEN * DMODEL) / 256, 256, 0, stream>>>(x, xb, S_LEN * DMODEL);
    transpose_cvt_kernel<<<(DMODEL * DMODEL) / 256, 256, 0, stream>>>(Wq, Wqt);
    transpose_cvt_kernel<<<(DMODEL * DMODEL) / 256, 256, 0, stream>>>(Wk, Wkt);
    transpose_cvt_kernel<<<(DMODEL * DMODEL) / 256, 256, 0, stream>>>(Wv, Wvt);
    transpose_cvt_kernel<<<(DMODEL * DMODEL) / 256, 256, 0, stream>>>(Wo, Wot);

    // 2) QKV projections (Q pre-scaled by 1/sqrt(dk))
    dim3 ggrid(DMODEL / BN, S_LEN / BM);   // (16, 32)
    gemm_bf16_kernel<<<ggrid, 256, 0, stream>>>(xb, Wqt, bq, Qh, nullptr,
                                                DMODEL, 0, 0.125f);
    gemm_bf16_kernel<<<ggrid, 256, 0, stream>>>(xb, Wkt, bk, Kh, nullptr,
                                                DMODEL, 1, 1.0f);
    gemm_bf16_kernel<<<ggrid, 256, 0, stream>>>(xb, Wvt, bv, Vt, nullptr,
                                                DMODEL, 2, 1.0f);

    // 3) flash attention: 32 q-blocks x 16 heads
    attn_kernel<<<dim3(S_LEN / 128, NHEADS), 256, 0, stream>>>(Qh, Kh, Vt, Ab);

    // 4) output projection -> fp32 d_out
    gemm_bf16_kernel<<<ggrid, 256, 0, stream>>>(Ab, Wot, bo, nullptr, out,
                                                DMODEL, 3, 1.0f);
}
